// Attention_47742856463089
// MI455X (gfx1250) — matrix operand order repaired and checked by reading, not compiled, not run on hardware
//
#include <hip/hip_runtime.h>

typedef unsigned int   u32t;
typedef unsigned short u16t;
typedef __attribute__((ext_vector_type(16))) __bf16 v16bf;
typedef __attribute__((ext_vector_type(8)))  float  v8f;
typedef int v4i_ __attribute__((vector_size(16)));   // matches builtin param type

#define B_   16
#define N_   1024
#define C_   1024
#define H_   16
#define D_   64
#define P_   16
#define M_   1040            // P + N keys
#define NC3  3072

// ---- optional CDNA5 async global->LDS path (guarded; sync fallback) -------
#if defined(__has_builtin)
#if __has_builtin(__builtin_amdgcn_global_load_async_to_lds_b128) && \
    __has_builtin(__builtin_amdgcn_s_wait_asynccnt)
#define HAVE_ASYNC_LDS 1
#endif
#endif
#ifndef HAVE_ASYNC_LDS
#define HAVE_ASYNC_LDS 0
#endif

#if HAVE_ASYNC_LDS
#define GLB_AS __attribute__((address_space(1)))
#define LDS_AS __attribute__((address_space(3)))
__device__ __forceinline__ void async_copy_b128(const void* g, void* l) {
  __builtin_amdgcn_global_load_async_to_lds_b128(
      (GLB_AS v4i_*)g, (LDS_AS v4i_*)l, 0, 0);
}
__device__ __forceinline__ void wait_async() {
  __builtin_amdgcn_s_wait_asynccnt(0);
}
#endif

union BFrag {
  u32t  u32[8];
  u16t  u16[16];
  v16bf v;
};

__device__ __forceinline__ u16t f2bf(float f) {
  u32t u = __float_as_uint(f);
  u += 0x7FFFu + ((u >> 16) & 1u);     // round-to-nearest-even
  return (u16t)(u >> 16);
}

// ---------------------------------------------------------------------------
// Kernel 0a: f32 -> bf16 bulk convert (vectorized x4)
// ---------------------------------------------------------------------------
__global__ __launch_bounds__(256) void convert_bf16(
    const float* __restrict__ src, u16t* __restrict__ dst, int n4) {
  int i = blockIdx.x * blockDim.x + threadIdx.x;
  if (i >= n4) return;
  float4 f = ((const float4*)src)[i];
  u32t p0 = (u32t)f2bf(f.x) | ((u32t)f2bf(f.y) << 16);
  u32t p1 = (u32t)f2bf(f.z) | ((u32t)f2bf(f.w) << 16);
  ((uint2*)dst)[i] = make_uint2(p0, p1);
}

// ---------------------------------------------------------------------------
// Kernel 0b: prefix pk/pv [B,P,C] -> K/V rows 0..P-1 in [B*H, M, D] bf16
// ---------------------------------------------------------------------------
__global__ __launch_bounds__(256) void prefix_kernel(
    const float* __restrict__ pk, const float* __restrict__ pv,
    u16t* __restrict__ Kb, u16t* __restrict__ Vb) {
  int e = blockIdx.x * blockDim.x + threadIdx.x;     // over B*P*C
  if (e >= B_ * P_ * C_) return;
  int b = e >> 14;               // P*C = 16384
  int rem = e & 16383;
  int p = rem >> 10;
  int c = rem & 1023;
  int h = c >> 6, d = c & 63;
  size_t dst = (((size_t)(b * H_ + h)) * M_ + p) * D_ + d;
  Kb[dst] = f2bf(pk[e]);
  Vb[dst] = f2bf(pv[e]);
}

// ---------------------------------------------------------------------------
// Kernel 1: QKV GEMM  [16384,1024](bf16) x [1024,3072](bf16) -> Q/K/V (bf16)
// 128x128 block tile, 256 threads = 8 waves, wave tile 32x64 (2x4 WMMA tiles)
// ---------------------------------------------------------------------------
__global__ __launch_bounds__(256) void gemm_qkv(
    const u16t* __restrict__ A,   // xb  [16384,1024]
    const u16t* __restrict__ Bw,  // wqb [1024,3072]
    u16t* __restrict__ Qb, u16t* __restrict__ Kb, u16t* __restrict__ Vb) {
  __shared__ __align__(16) u16t As[128 * 32];   // [row][k]
  __shared__ __align__(16) u16t Bs[128 * 32];   // [col][k] (transposed)

  const int tid  = threadIdx.x;
  const int lane = tid & 31;
  const int wv   = tid >> 5;
  const int wrow = wv & 3;
  const int wcol = wv >> 2;
  const int rowbase = blockIdx.y * 128;
  const int colbase = blockIdx.x * 128;

  v8f zero8 = {0.f, 0.f, 0.f, 0.f, 0.f, 0.f, 0.f, 0.f};
  v8f acc[2][4];
#pragma unroll
  for (int t = 0; t < 2; ++t)
#pragma unroll
    for (int u = 0; u < 4; ++u) acc[t][u] = zero8;

  const u32t* gA = (const u32t*)A;
  const u32t* gB = (const u32t*)Bw;

  for (int kk = 0; kk < 32; ++kk) {
    int k0 = kk * 32;
    // ---- stage A tile: 128 rows x 32 k = 512 x 16B chunks ----
#if HAVE_ASYNC_LDS
    {
      const char* gbase = (const char*)A;
#pragma unroll
      for (int i = 0; i < 2; ++i) {
        int e = tid * 2 + i;               // 0..511
        int row = e >> 2;                  // 4 x 16B per row k-slice
        int cp  = e & 3;
        async_copy_b128(gbase + (size_t)(rowbase + row) * 2048 + (size_t)k0 * 2 + cp * 16,
                        (char*)As + (size_t)e * 16);
      }
    }
#else
    {
      u32t* dA = (u32t*)As;
#pragma unroll
      for (int i = 0; i < 8; ++i) {
        int e = tid * 8 + i;
        int row = e >> 4;
        int kp  = e & 15;
        dA[e] = gA[(size_t)(rowbase + row) * 512 + (k0 >> 1) + kp];
      }
    }
#endif
    // ---- stage B tile transposed: 32 k x 128 cols ----
#pragma unroll
    for (int i = 0; i < 8; ++i) {
      int e = tid * 8 + i;
      int k  = e >> 6;          // 64 uints (128 cols) per k-row
      int cp = e & 63;
      u32t w = gB[(size_t)(k0 + k) * (NC3 / 2) + (colbase >> 1) + cp];
      Bs[(2 * cp)     * 32 + k] = (u16t)(w & 0xFFFFu);
      Bs[(2 * cp + 1) * 32 + k] = (u16t)(w >> 16);
    }
    if (kk < 31) {
      __builtin_prefetch(&gB[(size_t)(k0 + 32 + (tid & 31)) * (NC3 / 2) + (colbase >> 1)], 0, 3);
#if !HAVE_ASYNC_LDS
      __builtin_prefetch(&gA[(size_t)(rowbase + (tid >> 1)) * 512 + ((k0 + 32) >> 1)], 0, 3);
#endif
    }
#if HAVE_ASYNC_LDS
    wait_async();
#endif
    __syncthreads();

    const int rl    = lane & 15;
    const int khalf = (lane >> 4) * 8;
    const int kb    = (lane >> 4) * 8;
    BFrag af[2], bf[4];
#pragma unroll
    for (int t = 0; t < 2; ++t) {
      int row = wrow * 32 + t * 16 + rl;
#pragma unroll
      for (int j = 0; j < 8; ++j) {
        int k = khalf + (j < 4 ? 2 * j : 16 + 2 * (j - 4));
        af[t].u32[j] = *(const u32t*)&As[row * 32 + k];
      }
    }
#pragma unroll
    for (int u = 0; u < 4; ++u) {
      int col = wcol * 64 + u * 16 + rl;
#pragma unroll
      for (int j = 0; j < 8; ++j)
        bf[u].u32[j] = *(const u32t*)&Bs[col * 32 + kb + (j < 4 ? 2 * j : 16 + 2 * (j - 4))];
    }
#pragma unroll
    for (int t = 0; t < 2; ++t)
#pragma unroll
      for (int u = 0; u < 4; ++u)
        acc[t][u] = __builtin_amdgcn_wmma_f32_16x16x32_bf16(
            false, af[t].v, false, bf[u].v, (short)0, acc[t][u], false, false);
    __syncthreads();
  }

  // epilogue: scatter into Q/K/V [B*H, seq, 64] bf16
  const int rl     = lane & 15;
  const int rowoff = (lane >> 4) * 8;
#pragma unroll
  for (int t = 0; t < 2; ++t)
#pragma unroll
    for (int u = 0; u < 4; ++u)
#pragma unroll
      for (int r = 0; r < 8; ++r) {
        int grow = rowbase + wrow * 32 + t * 16 + rowoff + r;
        int gcol = colbase + wcol * 64 + u * 16 + rl;
        u16t bv = f2bf(acc[t][u][r]);
        int which = gcol >> 10;
        int rem   = gcol & 1023;
        int h = rem >> 6, d = rem & 63;
        int b = grow >> 10, n = grow & 1023;
        size_t bh = (size_t)(b * H_ + h);
        if (which == 0)      Qb[(bh * N_ + n) * D_ + d] = bv;
        else if (which == 1) Kb[(bh * M_ + P_ + n) * D_ + d] = bv;
        else                 Vb[(bh * M_ + P_ + n) * D_ + d] = bv;
      }
}

// ---------------------------------------------------------------------------
// Kernel 2: flash attention.  grid = (B*H, N/64), block = 128 (4 waves),
// each wave owns 16 q-rows; keys processed in chunks of 32.
// ---------------------------------------------------------------------------
__global__ __launch_bounds__(128) void attn_kernel(
    const u16t* __restrict__ Qb,   // [B*H, N, 64]
    const u16t* __restrict__ Kb,   // [B*H, M, 64]
    const u16t* __restrict__ Vb,   // [B*H, M, 64]
    u16t* __restrict__ Ob) {       // [B*N, C] bf16
  __shared__ __align__(16) u16t sK [32 * 64];   // [key][dim]
  __shared__ __align__(16) u16t sVT[64 * 32];   // [dim][key] (transposed)
  __shared__ __align__(16) u16t sP [4][16 * 32];

  const int lane = threadIdx.x & 31;
  const int wave = threadIdx.x >> 5;
  const int bh   = blockIdx.x;
  const int b = bh >> 4, h = bh & 15;
  const int qbase = blockIdx.y * 64 + wave * 16;

  const u16t* Qp = Qb + ((size_t)bh * N_ + qbase) * D_;
  const u16t* Kp = Kb + (size_t)bh * M_ * D_;
  const u32t* gV = (const u32t*)(Vb + (size_t)bh * M_ * D_);

  // Q fragments (A layout), head dim split into two 32-wide K chunks
  BFrag qf[2];
  {
    const u32t* Q32 = (const u32t*)Qp;
    int rowl  = lane & 15;
    int khalf = (lane >> 4) * 8;
#pragma unroll
    for (int c = 0; c < 2; ++c)
#pragma unroll
      for (int j = 0; j < 8; ++j) {
        int k = c * 32 + khalf + (j < 4 ? 2 * j : 16 + 2 * (j - 4));
        qf[c].u32[j] = Q32[rowl * 32 + (k >> 1)];
      }
  }

  v8f zero8 = {0.f, 0.f, 0.f, 0.f, 0.f, 0.f, 0.f, 0.f};
  float mrun[8], lrun[8];
  v8f oacc[4];
#pragma unroll
  for (int r = 0; r < 8; ++r) { mrun[r] = -1e30f; lrun[r] = 0.f; }
#pragma unroll
  for (int t = 0; t < 4; ++t) oacc[t] = zero8;

  const float scale = 0.125f;   // D^-0.5

  for (int kc = 0; kc < 33; ++kc) {
    const int keybase = kc * 32;
    // ---- stage K (row major) into LDS; zero-pad OOB keys ----
#if HAVE_ASYNC_LDS
    {
#pragma unroll
      for (int i = 0; i < 2; ++i) {
        int e = threadIdx.x * 2 + i;        // 0..255 x 16B chunks
        int key = e >> 3;                   // 8 x 16B per key row (128B)
        int cp  = e & 7;
        int kg  = keybase + key;
        if (kg < M_) {
          async_copy_b128((const char*)Kp + (size_t)kg * 128 + cp * 16,
                          (char*)sK + (size_t)e * 16);
        } else {
          int4 z = {0, 0, 0, 0};
          ((int4*)sK)[e] = z;
        }
      }
    }
#else
    {
      const u32t* gK = (const u32t*)Kp;
#pragma unroll
      for (int i = 0; i < 8; ++i) {
        int e = threadIdx.x * 8 + i;           // 0..1023 uints
        int key = e >> 5, dp = e & 31;
        int kg = keybase + key;
        ((u32t*)sK)[e] = (kg < M_) ? gK[(size_t)kg * 32 + dp] : 0u;
      }
    }
#endif
    // ---- stage V transposed into LDS ----
#pragma unroll
    for (int i = 0; i < 8; ++i) {
      int e = threadIdx.x * 8 + i;
      int key = e >> 5, dp = e & 31;
      int kg = keybase + key;
      u32t w = (kg < M_) ? gV[(size_t)kg * 32 + dp] : 0u;
      int d = dp * 2;
      sVT[d * 32 + key]       = (u16t)(w & 0xFFFFu);
      sVT[(d + 1) * 32 + key] = (u16t)(w >> 16);
    }
#if HAVE_ASYNC_LDS
    wait_async();
#endif
    __syncthreads();

    // ---- scores: two 16x16 tiles (keys t*16..t*16+15), K-dim = 64 ----
    v8f sacc[2];
#pragma unroll
    for (int t = 0; t < 2; ++t) {
      BFrag kb0, kb1;
      int keyl = t * 16 + (lane & 15);
      int krow = (lane >> 4) * 8;
#pragma unroll
      for (int j = 0; j < 8; ++j) {
        int d0 = krow + (j < 4 ? 2 * j : 16 + 2 * (j - 4));
        kb0.u32[j] = *(const u32t*)&sK[keyl * 64 + d0];
        kb1.u32[j] = *(const u32t*)&sK[keyl * 64 + 32 + d0];
      }
      v8f a = zero8;
      a = __builtin_amdgcn_wmma_f32_16x16x32_bf16(false, qf[0].v, false, kb0.v, (short)0, a, false, false);
      a = __builtin_amdgcn_wmma_f32_16x16x32_bf16(false, qf[1].v, false, kb1.v, (short)0, a, false, false);
      sacc[t] = a;
    }

    // ---- scale + OOB mask ----
#pragma unroll
    for (int t = 0; t < 2; ++t) {
      int kg = keybase + t * 16 + (lane & 15);
      bool valid = kg < M_;
#pragma unroll
      for (int r = 0; r < 8; ++r)
        sacc[t][r] = valid ? sacc[t][r] * scale : -1e30f;
    }

    // ---- online softmax; row stats reduce over 16-lane halves ----
    float alpha[8];
#pragma unroll
    for (int r = 0; r < 8; ++r) {
      float mx = fmaxf(sacc[0][r], sacc[1][r]);
#pragma unroll
      for (int off = 8; off >= 1; off >>= 1)
        mx = fmaxf(mx, __shfl_xor(mx, off, 32));
      float mnew = fmaxf(mrun[r], mx);
      float a    = __expf(mrun[r] - mnew);
      float p0   = __expf(sacc[0][r] - mnew);
      float p1   = __expf(sacc[1][r] - mnew);
      float rs   = p0 + p1;
#pragma unroll
      for (int off = 8; off >= 1; off >>= 1)
        rs += __shfl_xor(rs, off, 32);
      lrun[r] = lrun[r] * a + rs;
      mrun[r] = mnew;
      alpha[r] = a;
      sacc[0][r] = p0;
      sacc[1][r] = p1;
    }
#pragma unroll
    for (int t = 0; t < 4; ++t)
#pragma unroll
      for (int r = 0; r < 8; ++r) oacc[t][r] *= alpha[r];

    // ---- repack P (D layout -> A layout) via per-wave LDS ----
    {
      u16t* sp = sP[wave];
      int rowoff = (lane >> 4) * 8;
      int col0   = lane & 15;
#pragma unroll
      for (int t = 0; t < 2; ++t)
#pragma unroll
        for (int r = 0; r < 8; ++r)
          sp[(rowoff + r) * 32 + t * 16 + col0] = f2bf(sacc[t][r]);
    }
    BFrag pf;
    {
      const u16t* sp = sP[wave];
      int rowl  = lane & 15;
      int khalf = (lane >> 4) * 8;
#pragma unroll
      for (int j = 0; j < 8; ++j) {
        int k = khalf + (j < 4 ? 2 * j : 16 + 2 * (j - 4));
        pf.u32[j] = *(const u32t*)&sp[rowl * 32 + k];
      }
    }

    // ---- O += P x V over 4 output d-chunks ----
#pragma unroll
    for (int t = 0; t < 4; ++t) {
      BFrag vf;
      int dcol = t * 16 + (lane & 15);
      int krow = (lane >> 4) * 8;
#pragma unroll
      for (int j = 0; j < 8; ++j)
        vf.u32[j] = *(const u32t*)&sVT[dcol * 32 + krow + (j < 4 ? 2 * j : 16 + 2 * (j - 4))];
      oacc[t] = __builtin_amdgcn_wmma_f32_16x16x32_bf16(
          false, pf.v, false, vf.v, (short)0, oacc[t], false, false);
    }
    __syncthreads();
  }

  // ---- finalize: divide by l, store bf16 into [B*N, C] at col h*64+d ----
  float linv[8];
#pragma unroll
  for (int r = 0; r < 8; ++r) linv[r] = 1.0f / lrun[r];
  const int col0   = lane & 15;
  const int rowoff = (lane >> 4) * 8;
#pragma unroll
  for (int t = 0; t < 4; ++t)
#pragma unroll
    for (int r = 0; r < 8; ++r) {
      int row = qbase + rowoff + r;
      int d   = t * 16 + col0;
      Ob[((size_t)(b * N_ + row)) * C_ + h * D_ + d] = f2bf(oacc[t][r] * linv[r]);
    }
}

// ---------------------------------------------------------------------------
// Kernel 3: output projection [16384,1024](bf16) x [1024,1024](bf16) + bias
// -> f32 d_out.  Same tiling as gemm_qkv.
// ---------------------------------------------------------------------------
__global__ __launch_bounds__(256) void gemm_proj(
    const u16t* __restrict__ A,    // Ob  [16384,1024]
    const u16t* __restrict__ Bw,   // wpb [1024,1024]
    const float* __restrict__ bias,
    float* __restrict__ out) {
  __shared__ __align__(16) u16t As[128 * 32];
  __shared__ __align__(16) u16t Bs[128 * 32];

  const int tid  = threadIdx.x;
  const int lane = tid & 31;
  const int wv   = tid >> 5;
  const int wrow = wv & 3;
  const int wcol = wv >> 2;
  const int rowbase = blockIdx.y * 128;
  const int colbase = blockIdx.x * 128;

  v8f zero8 = {0.f, 0.f, 0.f, 0.f, 0.f, 0.f, 0.f, 0.f};
  v8f acc[2][4];
#pragma unroll
  for (int t = 0; t < 2; ++t)
#pragma unroll
    for (int u = 0; u < 4; ++u) acc[t][u] = zero8;

  const u32t* gA = (const u32t*)A;
  const u32t* gB = (const u32t*)Bw;

  for (int kk = 0; kk < 32; ++kk) {
    int k0 = kk * 32;
#if HAVE_ASYNC_LDS
    {
      const char* gbase = (const char*)A;
#pragma unroll
      for (int i = 0; i < 2; ++i) {
        int e = tid * 2 + i;
        int row = e >> 2;
        int cp  = e & 3;
        async_copy_b128(gbase + (size_t)(rowbase + row) * 2048 + (size_t)k0 * 2 + cp * 16,
                        (char*)As + (size_t)e * 16);
      }
    }
#else
    {
      u32t* dA = (u32t*)As;
#pragma unroll
      for (int i = 0; i < 8; ++i) {
        int e = tid * 8 + i;
        int row = e >> 4, kp = e & 15;
        dA[e] = gA[(size_t)(rowbase + row) * 512 + (k0 >> 1) + kp];
      }
    }
#endif
#pragma unroll
    for (int i = 0; i < 8; ++i) {
      int e = tid * 8 + i;
      int k = e >> 6, cp = e & 63;
      u32t w = gB[(size_t)(k0 + k) * (C_ / 2) + (colbase >> 1) + cp];
      Bs[(2 * cp)     * 32 + k] = (u16t)(w & 0xFFFFu);
      Bs[(2 * cp + 1) * 32 + k] = (u16t)(w >> 16);
    }
    if (kk < 31) {
      __builtin_prefetch(&gB[(size_t)(k0 + 32 + (tid & 31)) * (C_ / 2) + (colbase >> 1)], 0, 3);
#if !HAVE_ASYNC_LDS
      __builtin_prefetch(&gA[(size_t)(rowbase + (tid >> 1)) * 512 + ((k0 + 32) >> 1)], 0, 3);
#endif
    }
#if HAVE_ASYNC_LDS
    wait_async();
#endif
    __syncthreads();

    const int rl    = lane & 15;
    const int khalf = (lane >> 4) * 8;
    const int kb    = (lane >> 4) * 8;
    BFrag af[2], bf[4];
#pragma unroll
    for (int t = 0; t < 2; ++t) {
      int row = wrow * 32 + t * 16 + rl;
#pragma unroll
      for (int j = 0; j < 8; ++j) {
        int k = khalf + (j < 4 ? 2 * j : 16 + 2 * (j - 4));
        af[t].u32[j] = *(const u32t*)&As[row * 32 + k];
      }
    }
#pragma unroll
    for (int u = 0; u < 4; ++u) {
      int col = wcol * 64 + u * 16 + rl;
#pragma unroll
      for (int j = 0; j < 8; ++j)
        bf[u].u32[j] = *(const u32t*)&Bs[col * 32 + kb + (j < 4 ? 2 * j : 16 + 2 * (j - 4))];
    }
#pragma unroll
    for (int t = 0; t < 2; ++t)
#pragma unroll
      for (int u = 0; u < 4; ++u)
        acc[t][u] = __builtin_amdgcn_wmma_f32_16x16x32_bf16(
            false, af[t].v, false, bf[u].v, (short)0, acc[t][u], false, false);
    __syncthreads();
  }

  const int rl     = lane & 15;
  const int rowoff = (lane >> 4) * 8;
#pragma unroll
  for (int t = 0; t < 2; ++t)
#pragma unroll
    for (int u = 0; u < 4; ++u) {
      int gcol = colbase + wcol * 64 + u * 16 + rl;
      float bb = bias[gcol];
#pragma unroll
      for (int r = 0; r < 8; ++r) {
        int grow = rowbase + wrow * 32 + t * 16 + rowoff + r;
        out[(size_t)grow * C_ + gcol] = acc[t][u][r] + bb;
      }
    }
}

// ---------------------------------------------------------------------------
extern "C" void kernel_launch(void* const* d_in, const int* in_sizes, int n_in,
                              void* d_out, int out_size, void* d_ws, size_t ws_size,
                              hipStream_t stream) {
  const float* x      = (const float*)d_in[0];
  const float* pk     = (const float*)d_in[1];
  const float* pv     = (const float*)d_in[2];
  const float* w_qkv  = (const float*)d_in[3];
  const float* w_proj = (const float*)d_in[4];
  const float* b_proj = (const float*)d_in[5];
  float* out = (float*)d_out;

  char* ws = (char*)d_ws;
  size_t off = 0;
  auto alloc = [&](size_t bytes) -> void* {
    void* p = ws + off;
    off = (off + bytes + 255) & ~(size_t)255;
    return p;
  };
  u16t* xb  = (u16t*)alloc((size_t)B_ * N_ * C_ * 2);
  u16t* wqb = (u16t*)alloc((size_t)C_ * NC3 * 2);
  u16t* wpb = (u16t*)alloc((size_t)C_ * C_ * 2);
  u16t* Qb  = (u16t*)alloc((size_t)B_ * H_ * N_ * D_ * 2);
  u16t* Kb  = (u16t*)alloc((size_t)B_ * H_ * M_ * D_ * 2);
  u16t* Vb  = (u16t*)alloc((size_t)B_ * H_ * M_ * D_ * 2);
  u16t* Ob  = (u16t*)alloc((size_t)B_ * N_ * C_ * 2);

  // 1) precision conversion (f32 -> bf16)
  convert_bf16<<<dim3((B_ * N_ * C_ / 4) / 256), dim3(256), 0, stream>>>(x, xb, B_ * N_ * C_ / 4);
  convert_bf16<<<dim3((C_ * NC3 / 4) / 256), dim3(256), 0, stream>>>(w_qkv, wqb, C_ * NC3 / 4);
  convert_bf16<<<dim3((C_ * C_ / 4) / 256), dim3(256), 0, stream>>>(w_proj, wpb, C_ * C_ / 4);
  prefix_kernel<<<dim3((B_ * P_ * C_) / 256), dim3(256), 0, stream>>>(pk, pv, Kb, Vb);

  // 2) QKV projection GEMM
  gemm_qkv<<<dim3(NC3 / 128, (B_ * N_) / 128), dim3(256), 0, stream>>>(xb, wqb, Qb, Kb, Vb);

  // 3) flash attention
  attn_kernel<<<dim3(B_ * H_, N_ / 64), dim3(128), 0, stream>>>(Qb, Kb, Vb, Ob);

  // 4) output projection + bias
  gemm_proj<<<dim3(C_ / 128, (B_ * N_) / 128), dim3(256), 0, stream>>>(Ob, wpb, b_proj, out);

  (void)in_sizes; (void)n_in; (void)out_size; (void)ws_size;
}
